// GIN_MOD_16123307229971
// MI455X (gfx1250) — matrix operand order repaired and checked by reading, not compiled, not run on hardware
//
#include <hip/hip_runtime.h>

typedef __attribute__((ext_vector_type(16))) _Float16 v16h;
typedef __attribute__((ext_vector_type(8)))  _Float16 v8h;
typedef __attribute__((ext_vector_type(8)))  float    v8f;

#define D_DIM   300
#define DH_DIM  600
#define KPAD1   320          // K for GEMM1, padded to mult of 32
#define KT1     10           // KPAD1/32
#define NT1     38           // 608/16 output tiles (600 padded to 608)
#define TPAD    608          // t (hidden) row stride, mult of 32
#define KT2     19           // TPAD/32
#define NT2     19           // 304/16 output tiles (300 padded)
#define NLAYERS 5
#define BN_EPS  1e-5f

__device__ __forceinline__ void atomAddF(float* p, float v) {
  unsafeAtomicAdd(p, v);   // global_atomic_add_f32 (no CAS loop)
}

__device__ __forceinline__ void waitAsync0() {
#if __has_builtin(__builtin_amdgcn_s_wait_asynccnt)
  __builtin_amdgcn_s_wait_asynccnt(0);
#else
  asm volatile("s_wait_asynccnt 0x0" ::: "memory");
#endif
}

// ---------------------------------------------------------------------------
// h[n, :] = emb_atom[atomic_number[n]] + emb_chir[chirality[n]]
// ---------------------------------------------------------------------------
__global__ void embed_kernel(const int* __restrict__ an, const int* __restrict__ ct,
                             const float* __restrict__ ea, const float* __restrict__ ec,
                             float* __restrict__ h, int N) {
  long long i = (long long)blockIdx.x * blockDim.x + threadIdx.x;
  long long total = (long long)N * D_DIM;
  if (i >= total) return;
  int n = (int)(i / D_DIM);
  int c = (int)(i % D_DIM);
  h[i] = ea[(size_t)an[n] * D_DIM + c] + ec[(size_t)ct[n] * D_DIM + c];
}

// ---------------------------------------------------------------------------
// Repack W1 (f32 [L][300][600]) into per-lane WMMA B-fragment layout, f16.
// Fragment element hj (0..15) of lane L = W[k = kt*32 + (L>>4)*16 + hj]
//                                          [col = nt*16 + (L&15)]
// ---------------------------------------------------------------------------
__global__ void pack_w1_kernel(const float* __restrict__ W1, _Float16* __restrict__ Wp) {
  int idx = blockIdx.x * blockDim.x + threadIdx.x;
  const int total = NLAYERS * NT1 * KT1 * 32 * 16;
  if (idx >= total) return;
  int hj   = idx & 15;
  int lane = (idx >> 4) & 31;
  int rest = idx >> 9;
  int kt = rest % KT1;  rest /= KT1;
  int nt = rest % NT1;
  int l  = rest / NT1;
  int k   = kt * 32 + ((lane >> 4) << 4) + hj;
  int col = nt * 16 + (lane & 15);
  float v = 0.f;
  if (k < D_DIM && col < DH_DIM)
    v = W1[((size_t)l * D_DIM + k) * DH_DIM + col];
  Wp[idx] = (_Float16)v;
}

__global__ void pack_w2_kernel(const float* __restrict__ W2, _Float16* __restrict__ Wp) {
  int idx = blockIdx.x * blockDim.x + threadIdx.x;
  const int total = NLAYERS * NT2 * KT2 * 32 * 16;
  if (idx >= total) return;
  int hj   = idx & 15;
  int lane = (idx >> 4) & 31;
  int rest = idx >> 9;
  int kt = rest % KT2;  rest /= KT2;
  int nt = rest % NT2;
  int l  = rest / NT2;
  int k   = kt * 32 + ((lane >> 4) << 4) + hj;
  int col = nt * 16 + (lane & 15);
  float v = 0.f;
  if (k < DH_DIM && col < D_DIM)
    v = W2[((size_t)l * DH_DIM + k) * D_DIM + col];
  Wp[idx] = (_Float16)v;
}

// ---------------------------------------------------------------------------
// One wave32 per edge: agg[dst] += h[src] + eb[bond] + ed[dir]
// ---------------------------------------------------------------------------
__global__ void scatter_kernel(const float* __restrict__ h,
                               const int* __restrict__ src, const int* __restrict__ dst,
                               const int* __restrict__ bt, const int* __restrict__ bd,
                               const float* __restrict__ eb, const float* __restrict__ ed,
                               float* __restrict__ agg, int E) {
  long long t = (long long)blockIdx.x * blockDim.x + threadIdx.x;
  int e    = (int)(t >> 5);
  int lane = (int)(t & 31);
  if (e >= E) return;
  int s = src[e], d = dst[e], b = bt[e], dir = bd[e];
  const float* hs  = h  + (size_t)s   * D_DIM;
  const float* ebp = eb + (size_t)b   * D_DIM;
  const float* edp = ed + (size_t)dir * D_DIM;
  float*       ap  = agg + (size_t)d  * D_DIM;
  for (int c = lane; c < D_DIM; c += 32)
    atomAddF(ap + c, hs[c] + ebp[c] + edp[c]);
}

// ---------------------------------------------------------------------------
// Build A fragment (16x32 f16) from LDS row-major tile.
// Lane L: row m = L&15; kb = (L>>4)*8; elements = As[m][k0+2*kb..+15]
// ---------------------------------------------------------------------------
__device__ __forceinline__ v16h load_a_frag(const _Float16* __restrict__ As,
                                            int kpad, int m, int kb, int k0) {
  const _Float16* ap = As + (size_t)m * kpad + k0 + 2 * kb;
  v8h lo = *(const v8h*)ap;
  v8h hi = *(const v8h*)(ap + 8);
  return __builtin_shufflevector(lo, hi, 0,1,2,3,4,5,6,7,8,9,10,11,12,13,14,15);
}

// ---------------------------------------------------------------------------
// GEMM1: t[16 rows x 608] = relu((h+agg) @ W1 + b1), f16 output
// Block = 128 threads (4 waves), one 16-row tile per block, waves split N-tiles.
// Staging is VALU (fuses h+agg and f32->f16 conversion).
// ---------------------------------------------------------------------------
__global__ __launch_bounds__(128) void gemm1_kernel(
    const float* __restrict__ h, const float* __restrict__ agg,
    const _Float16* __restrict__ w1p, const float* __restrict__ b1,
    _Float16* __restrict__ tout, int N) {
  __shared__ _Float16 As[16 * KPAD1];
  const int rowbase = blockIdx.x * 16;
  for (int i = threadIdx.x; i < 16 * (KPAD1 / 4); i += 128) {
    int r  = i / (KPAD1 / 4);
    int kc = (i % (KPAD1 / 4)) * 4;
    int row = rowbase + r;
    float4 v = make_float4(0.f, 0.f, 0.f, 0.f);
    if (row < N && kc < D_DIM) {
      float4 a = *(const float4*)(h   + (size_t)row * D_DIM + kc);
      float4 g = *(const float4*)(agg + (size_t)row * D_DIM + kc);
      v = make_float4(a.x + g.x, a.y + g.y, a.z + g.z, a.w + g.w);
    }
    _Float16* dp = &As[r * KPAD1 + kc];
    dp[0] = (_Float16)v.x; dp[1] = (_Float16)v.y;
    dp[2] = (_Float16)v.z; dp[3] = (_Float16)v.w;
  }
  __syncthreads();
  const int wave = threadIdx.x >> 5, lane = threadIdx.x & 31;
  const int m    = lane & 15;
  const int kb   = (lane >> 4) * 8;
  const int ncol = lane & 15;
  const int mrow0 = rowbase + ((lane >> 4) * 8);
  for (int nt = wave; nt < NT1; nt += 4) {
    v8f acc = {};
    for (int kt = 0; kt < KT1; ++kt) {
      v16h a = load_a_frag(As, KPAD1, m, kb, kt * 32);
      v16h b = *(const v16h*)(w1p + ((size_t)(nt * KT1 + kt) * 32 + lane) * 16);
      acc = __builtin_amdgcn_wmma_f32_16x16x32_f16(false, a, false, b,
                                                   (short)0, acc, false, false);
    }
    int col = nt * 16 + ncol;
    float bias = (col < DH_DIM) ? b1[col] : 0.f;
    for (int r = 0; r < 8; ++r) {
      int row = mrow0 + r;
      if (row < N) {
        float v = acc[r] + bias;
        v = v > 0.f ? v : 0.f;                       // ReLU
        tout[(size_t)row * TPAD + col] = (_Float16)v;
      }
    }
  }
}

// ---------------------------------------------------------------------------
// GEMM2: h[16 rows x 300] = BN(t @ W2 + b2) (+ReLU if do_relu), f32 output
// A-stage uses CDNA5 async global->LDS copy (ASYNCcnt-tracked DMA path).
// ---------------------------------------------------------------------------
__global__ __launch_bounds__(128) void gemm2_kernel(
    const _Float16* __restrict__ tin, const _Float16* __restrict__ w2p,
    const float* __restrict__ b2,
    const float* __restrict__ gamma, const float* __restrict__ beta,
    const float* __restrict__ mean,  const float* __restrict__ var,
    float* __restrict__ h, int N, int do_relu) {
  __shared__ _Float16 As[16 * TPAD];
  const int rowbase = blockIdx.x * 16;
  {
    unsigned ldsbase = (unsigned)(uintptr_t)(&As[0]);   // shared aperture: low 32 bits = LDS addr
    for (int i = threadIdx.x; i < 16 * (TPAD / 8); i += 128) {
      int r  = i / (TPAD / 8);
      int kc = (i % (TPAD / 8)) * 8;
      int row = rowbase + r;
      if (row < N) {
        const _Float16* gp = tin + (size_t)row * TPAD + kc;
        unsigned la = ldsbase + (unsigned)((r * TPAD + kc) * 2);
        // async 16B global->LDS copy, tracked on ASYNCcnt
        asm volatile("global_load_async_to_lds_b128 %0, %1, off"
                     :: "v"(la), "v"(gp) : "memory");
      } else {
        *(v8h*)&As[r * TPAD + kc] = (v8h){};            // zero-pad tail rows
      }
    }
    waitAsync0();
  }
  __syncthreads();
  const int wave = threadIdx.x >> 5, lane = threadIdx.x & 31;
  const int m    = lane & 15;
  const int kb   = (lane >> 4) * 8;
  const int ncol = lane & 15;
  const int mrow0 = rowbase + ((lane >> 4) * 8);
  for (int nt = wave; nt < NT2; nt += 4) {
    v8f acc = {};
    for (int kt = 0; kt < KT2; ++kt) {
      v16h a = load_a_frag(As, TPAD, m, kb, kt * 32);
      v16h b = *(const v16h*)(w2p + ((size_t)(nt * KT2 + kt) * 32 + lane) * 16);
      acc = __builtin_amdgcn_wmma_f32_16x16x32_f16(false, a, false, b,
                                                   (short)0, acc, false, false);
    }
    int col = nt * 16 + ncol;
    if (col < D_DIM) {
      float sc = gamma[col] * rsqrtf(var[col] + BN_EPS);
      float mn = mean[col], be = beta[col], bb = b2[col];
      for (int r = 0; r < 8; ++r) {
        int row = mrow0 + r;
        if (row < N) {
          float z = acc[r] + bb;
          z = (z - mn) * sc + be;
          if (do_relu) z = fmaxf(z, 0.f);
          h[(size_t)row * D_DIM + col] = z;
        }
      }
    }
  }
}

// ---------------------------------------------------------------------------
// One wave32 per node: sums[gid[n]] += h[n]; cnt[gid[n]] += 1
// ---------------------------------------------------------------------------
__global__ void pool_kernel(const float* __restrict__ h, const int* __restrict__ gid,
                            float* __restrict__ sums, float* __restrict__ cnt, int N) {
  long long t = (long long)blockIdx.x * blockDim.x + threadIdx.x;
  int n    = (int)(t >> 5);
  int lane = (int)(t & 31);
  if (n >= N) return;
  int g = gid[n];
  const float* hp = h + (size_t)n * D_DIM;
  float* sp = sums + (size_t)g * 304;
  for (int c = lane; c < D_DIM; c += 32) atomAddF(sp + c, hp[c]);
  if (lane == 0) atomAddF(cnt + g, 1.0f);
}

// ---------------------------------------------------------------------------
// Head: x = sums/cnt; y1 = x@Wt+bt; y2 = leaky(y1@Wh+bh); out = y2@Wf+bf
// One block (256 thr) per graph.
// ---------------------------------------------------------------------------
__global__ __launch_bounds__(256) void head_kernel(
    const float* __restrict__ sums, const float* __restrict__ cnt,
    const float* __restrict__ Wt, const float* __restrict__ bt,
    const float* __restrict__ Wh, const float* __restrict__ bh,
    const float* __restrict__ Wf, const float* __restrict__ bf,
    float* __restrict__ out) {
  __shared__ float xs[304];
  __shared__ float y1[256];
  __shared__ float y2[128];
  int g = blockIdx.x, tid = threadIdx.x;
  float invc = 1.0f / fmaxf(cnt[g], 1.0f);
  for (int c = tid; c < D_DIM; c += 256) xs[c] = sums[(size_t)g * 304 + c] * invc;
  __syncthreads();
  float acc = bt[tid];
  for (int k = 0; k < D_DIM; ++k) acc += xs[k] * Wt[(size_t)k * 256 + tid];
  y1[tid] = acc;
  __syncthreads();
  if (tid < 128) {
    float a2 = bh[tid];
    for (int k = 0; k < 256; ++k) a2 += y1[k] * Wh[(size_t)k * 128 + tid];
    y2[tid] = a2 > 0.f ? a2 : 0.01f * a2;   // leaky_relu, slope 0.01
  }
  __syncthreads();
  if (tid < 128) y1[tid] = y2[tid] * Wf[tid];
  __syncthreads();
  for (int s = 64; s > 0; s >>= 1) {
    if (tid < s) y1[tid] += y1[tid + s];
    __syncthreads();
  }
  if (tid == 0) out[g] = y1[0] + bf[0];
}

// ---------------------------------------------------------------------------
extern "C" void kernel_launch(void* const* d_in, const int* in_sizes, int n_in,
                              void* d_out, int out_size, void* d_ws, size_t ws_size,
                              hipStream_t stream) {
  const int*   atomic_number = (const int*)d_in[0];
  const int*   chirality     = (const int*)d_in[1];
  const int*   bond_type     = (const int*)d_in[2];
  const int*   bond_dir      = (const int*)d_in[3];
  const int*   src           = (const int*)d_in[4];
  const int*   dst           = (const int*)d_in[5];
  const int*   graph_id      = (const int*)d_in[6];
  const float* emb_atom      = (const float*)d_in[8];
  const float* emb_chir      = (const float*)d_in[9];
  const float* edge_bond     = (const float*)d_in[10];
  const float* edge_dir      = (const float*)d_in[11];
  const float* W1            = (const float*)d_in[12];
  const float* b1            = (const float*)d_in[13];
  const float* W2            = (const float*)d_in[14];
  const float* b2            = (const float*)d_in[15];
  const float* bn_gamma      = (const float*)d_in[16];
  const float* bn_beta       = (const float*)d_in[17];
  const float* bn_mean       = (const float*)d_in[18];
  const float* bn_var        = (const float*)d_in[19];
  const float* Wt            = (const float*)d_in[20];
  const float* bt            = (const float*)d_in[21];
  const float* Wh            = (const float*)d_in[22];
  const float* bh            = (const float*)d_in[23];
  const float* Wf            = (const float*)d_in[24];
  const float* bf            = (const float*)d_in[25];
  float* out = (float*)d_out;

  const int N = in_sizes[0];
  const int E = in_sizes[2];
  const int G = out_size;              // OUT_DIM == 1

  // workspace carve-out (256B aligned)
  char* ws = (char*)d_ws;
  size_t off = 0;
  auto carve = [&](size_t bytes) -> void* {
    void* p = ws + off;
    off += (bytes + 255) & ~(size_t)255;
    return p;
  };
  float*    h    = (float*)   carve((size_t)N * D_DIM * sizeof(float));
  float*    agg  = (float*)   carve((size_t)N * D_DIM * sizeof(float));
  _Float16* tbuf = (_Float16*)carve((size_t)N * TPAD * sizeof(_Float16));
  float*    sums = (float*)   carve((size_t)G * 304 * sizeof(float));
  float*    cnt  = (float*)   carve((size_t)G * sizeof(float));
  _Float16* w1p  = (_Float16*)carve((size_t)NLAYERS * NT1 * KT1 * 512 * sizeof(_Float16));
  _Float16* w2p  = (_Float16*)carve((size_t)NLAYERS * NT2 * KT2 * 512 * sizeof(_Float16));
  if (off > ws_size) return;

  // pack weights (tiny, once per call)
  {
    int tot1 = NLAYERS * NT1 * KT1 * 512;
    int tot2 = NLAYERS * NT2 * KT2 * 512;
    pack_w1_kernel<<<(tot1 + 255) / 256, 256, 0, stream>>>(W1, w1p);
    pack_w2_kernel<<<(tot2 + 255) / 256, 256, 0, stream>>>(W2, w2p);
  }

  // input embedding
  {
    long long tot = (long long)N * D_DIM;
    embed_kernel<<<(unsigned)((tot + 255) / 256), 256, 0, stream>>>(
        atomic_number, chirality, emb_atom, emb_chir, h, N);
  }

  const unsigned gemmGrid    = (unsigned)((N + 15) / 16);
  const unsigned scatterGrid = (unsigned)(((long long)E * 32 + 255) / 256);
  const unsigned poolGrid    = (unsigned)(((long long)N * 32 + 255) / 256);

  for (int l = 0; l < NLAYERS; ++l) {
    hipMemsetAsync(agg, 0, (size_t)N * D_DIM * sizeof(float), stream);
    scatter_kernel<<<scatterGrid, 256, 0, stream>>>(
        h, src, dst, bond_type, bond_dir,
        edge_bond + (size_t)l * 6 * D_DIM,
        edge_dir  + (size_t)l * 3 * D_DIM, agg, E);
    gemm1_kernel<<<gemmGrid, 128, 0, stream>>>(
        h, agg, w1p + (size_t)l * NT1 * KT1 * 512, b1 + (size_t)l * DH_DIM, tbuf, N);
    gemm2_kernel<<<gemmGrid, 128, 0, stream>>>(
        tbuf, w2p + (size_t)l * NT2 * KT2 * 512, b2 + (size_t)l * D_DIM,
        bn_gamma + (size_t)l * D_DIM, bn_beta + (size_t)l * D_DIM,
        bn_mean  + (size_t)l * D_DIM, bn_var  + (size_t)l * D_DIM,
        h, N, (l < NLAYERS - 1) ? 1 : 0);
  }

  // pooling + head
  hipMemsetAsync(sums, 0, (size_t)G * 304 * sizeof(float), stream);
  hipMemsetAsync(cnt,  0, (size_t)G * sizeof(float), stream);
  pool_kernel<<<poolGrid, 256, 0, stream>>>(h, graph_id, sums, cnt, N);
  head_kernel<<<G, 256, 0, stream>>>(sums, cnt, Wt, bt, Wh, bh, Wf, bf, out);
}
